// WCL_87522843558218
// MI455X (gfx1250) — matrix operand order repaired and checked by reading, not compiled, not run on hardware
//
#include <hip/hip_runtime.h>

typedef __attribute__((ext_vector_type(16))) _Float16 v16h;
typedef __attribute__((ext_vector_type(8)))  float    v8f;

// ---------------------------------------------------------------------------
// CDNA5 async copy: 16 bytes global -> LDS per lane, tracked by ASYNCcnt.
// ---------------------------------------------------------------------------
__device__ __forceinline__ void async_ld16(unsigned lds, const _Float16* g) {
  asm volatile("global_load_async_to_lds_b128 %0, %1, off"
               :: "v"(lds), "v"(g)
               : "memory");
}

// ---------------------------------------------------------------------------
// f32 -> f16 conversion, 4 elements per thread
// ---------------------------------------------------------------------------
__global__ __launch_bounds__(256)
void cvt_f32_to_f16(const float* __restrict__ in, _Float16* __restrict__ out, size_t total4) {
  size_t q = (size_t)blockIdx.x * 256 + threadIdx.x;
  if (q >= total4) return;
  size_t base = q * 4;
  float4 v = *reinterpret_cast<const float4*>(in + base);
  union { _Float16 h[4]; uint2 u; } pk;
  pk.h[0] = (_Float16)v.x; pk.h[1] = (_Float16)v.y;
  pk.h[2] = (_Float16)v.z; pk.h[3] = (_Float16)v.w;
  *reinterpret_cast<uint2*>(out + base) = pk.u;
}

// ---------------------------------------------------------------------------
// C[M,N] = A[M,K] * B[N,K]^T (+ bias[N]), f16 in / f32 out.
// Block: 256 threads (8 waves, 4x2). Block tile 256x128, wave tile 64x64,
// BK=32 -> 16 x v_wmma_f32_16x16x32_f16 per wave per K-step.
// LDS double-buffered, filled by per-lane global_load_async_to_lds_b128 with
// tiles pre-arranged in WMMA fragment layout:
//   A lane l (h=l>>4, m=l&15): element i -> K = (i&7) + h*8 + (i>=8)*16
//   B lane l (h=l>>4, n=l&15): element i -> K = (i&7) + h*8 + (i>=8)*16
// Main K-loop is peeled so the hot body is branch-free.
// ---------------------------------------------------------------------------
__global__ __launch_bounds__(256)
void gemm_nt_wmma(const _Float16* __restrict__ A, const _Float16* __restrict__ Bm,
                  float* __restrict__ C, const float* __restrict__ bias,
                  int M, int N, int K) {
  // per buffer: A = 16 subtiles * 32 lanes * 16 halfs = 8192 halfs (16KB)
  //             B =  8 subtiles * 32 lanes * 16 halfs = 4096 halfs ( 8KB)
  __shared__ __align__(32) _Float16 sMem[2 * 12288];   // 48 KB of 320 KB WGP LDS

  const int tid     = threadIdx.x;
  const int rowBase = blockIdx.y * 256;
  const int colBase = blockIdx.x * 128;

  // ---- staging roles: A slots tid and tid+256, B slot tid (slot = subtile*32+lane)
  const int a0_sub = tid >> 5,         a0_ln = tid & 31;
  const int a1_sub = (tid + 256) >> 5, a1_ln = tid & 31;
  const int b_sub  = tid >> 5,         b_ln  = tid & 31;

  const _Float16* aRow0 = A  + (size_t)(rowBase + a0_sub * 16 + (a0_ln & 15)) * K + ((a0_ln >> 4) * 8);
  const _Float16* aRow1 = A  + (size_t)(rowBase + a1_sub * 16 + (a1_ln & 15)) * K + ((a1_ln >> 4) * 8);
  const _Float16* bRow  = Bm + (size_t)(colBase + b_sub  * 16 + (b_ln  & 15)) * K + ((b_ln  >> 4) * 8);

  const unsigned ldsA0 = (unsigned)(uintptr_t)&sMem[(a0_sub * 32 + a0_ln) * 16];
  const unsigned ldsA1 = (unsigned)(uintptr_t)&sMem[(a1_sub * 32 + a1_ln) * 16];
  const unsigned ldsB  = (unsigned)(uintptr_t)&sMem[8192 + (b_sub * 32 + b_ln) * 16];

  auto issue = [&](int k0, int buf) {
    unsigned bo = (unsigned)buf * 24576u;           // buffer stride in bytes
    async_ld16(ldsA0 + bo,       aRow0 + k0);       // A slot0, elems 0..7
    async_ld16(ldsA0 + bo + 16u, aRow0 + k0 + 16);  // A slot0, elems 8..15
    async_ld16(ldsA1 + bo,       aRow1 + k0);
    async_ld16(ldsA1 + bo + 16u, aRow1 + k0 + 16);
    async_ld16(ldsB  + bo,       bRow  + k0);       // B slot, elems 0..7
    async_ld16(ldsB  + bo + 16u, bRow  + k0 + 16);  // B slot, elems 8..15
  };

  // ---- compute roles: wave -> 64x64 region (4 x 4 subtiles)
  const int w    = tid >> 5;
  const int lane = tid & 31;
  const int wm   = w >> 1;   // 0..3
  const int wn   = w & 1;    // 0..1

  const v8f vzero = {0.f, 0.f, 0.f, 0.f, 0.f, 0.f, 0.f, 0.f};
  v8f acc[4][4];
#pragma unroll
  for (int im = 0; im < 4; ++im)
#pragma unroll
    for (int in = 0; in < 4; ++in) acc[im][in] = vzero;

  auto compute = [&](int buf) {
    const _Float16* base = &sMem[buf * 12288];
    v16h af[4], bf[4];
#pragma unroll
    for (int im = 0; im < 4; ++im)
      af[im] = *reinterpret_cast<const v16h*>(&base[((wm * 4 + im) * 32 + lane) * 16]);
#pragma unroll
    for (int in = 0; in < 4; ++in)
      bf[in] = *reinterpret_cast<const v16h*>(&base[8192 + ((wn * 4 + in) * 32 + lane) * 16]);
#pragma unroll
    for (int im = 0; im < 4; ++im)
#pragma unroll
      for (int in = 0; in < 4; ++in)
        acc[im][in] = __builtin_amdgcn_wmma_f32_16x16x32_f16(
            false, af[im], false, bf[in], (short)0, acc[im][in], false, false);
  };

  issue(0, 0);                                      // prologue prefetch
  int cur = 0;
  int k0 = 0;
  // main loop (K/32 - 1 iterations): branch-free body
  for (; k0 + 32 < K; k0 += 32) {
    issue(k0 + 32, cur ^ 1);                        // prefetch next buffer
    asm volatile("s_wait_asynccnt 0x6" ::: "memory");    // current 6 done (in-order)
    __syncthreads();                                // all waves' current tile visible
    compute(cur);
    __syncthreads();                                // done reading cur -> writable next iter
    cur ^= 1;
  }
  // peeled last K-step: no prefetch
  asm volatile("s_wait_asynccnt 0x0" ::: "memory");
  __syncthreads();
  compute(cur);

  // epilogue: C/D layout — VGPR r: lanes 0-15 -> M=r, lanes 16-31 -> M=r+8
  const int hl  = lane >> 4;
  const int n16 = lane & 15;
#pragma unroll
  for (int im = 0; im < 4; ++im) {
#pragma unroll
    for (int in = 0; in < 4; ++in) {
      int col = colBase + (wn * 4 + in) * 16 + n16;
      float bv = bias ? bias[col] : 0.f;
#pragma unroll
      for (int r = 0; r < 8; ++r) {
        int row = rowBase + (wm * 4 + im) * 16 + hl * 8 + r;
        C[(size_t)row * N + col] = acc[im][in][r] + bv;
      }
    }
  }
}

// ---------------------------------------------------------------------------
// BatchNorm batch statistics, two-stage (deterministic, no float atomics).
// Stage 1: partial sums over 256-row chunks. Stage 2: finalize mu/rstd.
// ---------------------------------------------------------------------------
__global__ __launch_bounds__(256)
void bn_partial(const float* __restrict__ h, float* __restrict__ ps,
                float* __restrict__ pss, int Hn) {
  int c  = blockIdx.x * 256 + threadIdx.x;
  int rb = blockIdx.y;
  float s = 0.f, ss = 0.f;
  for (int r = rb * 256; r < rb * 256 + 256; ++r) {
    float v = h[(size_t)r * Hn + c];
    s += v; ss += v * v;
  }
  ps [(size_t)rb * Hn + c] = s;
  pss[(size_t)rb * Hn + c] = ss;
}

__global__ __launch_bounds__(256)
void bn_finalize(const float* __restrict__ ps, const float* __restrict__ pss,
                 float* __restrict__ mu, float* __restrict__ rstd,
                 int chunks, int Bn, int Hn) {
  int c = blockIdx.x * 256 + threadIdx.x;
  if (c >= Hn) return;
  float s = 0.f, ss = 0.f;
  for (int k = 0; k < chunks; ++k) {
    s  += ps [(size_t)k * Hn + c];
    ss += pss[(size_t)k * Hn + c];
  }
  float m   = s / (float)Bn;
  float var = ss / (float)Bn - m * m;    // biased variance
  mu[c]   = m;
  rstd[c] = rsqrtf(var + 1e-5f);
}

// ---------------------------------------------------------------------------
// BN apply + ReLU, emit f16 for next GEMM
// ---------------------------------------------------------------------------
__global__ __launch_bounds__(256)
void bn_relu_f16(const float* __restrict__ h, const float* __restrict__ mu,
                 const float* __restrict__ rstd, const float* __restrict__ gamma,
                 const float* __restrict__ beta, _Float16* __restrict__ o,
                 int Hn, size_t total4) {
  size_t q = (size_t)blockIdx.x * 256 + threadIdx.x;
  if (q >= total4) return;
  size_t base = q * 4;
  int c = (int)(base % Hn);
  float4 v = *reinterpret_cast<const float4*>(h + base);
  float vv[4] = {v.x, v.y, v.z, v.w};
  union { _Float16 h4[4]; uint2 u; } pk;
#pragma unroll
  for (int k = 0; k < 4; ++k) {
    int cc = c + k;
    float t = (vv[k] - mu[cc]) * rstd[cc] * gamma[cc] + beta[cc];
    pk.h4[k] = (_Float16)fmaxf(t, 0.f);
  }
  *reinterpret_cast<uint2*>(o + base) = pk.u;
}

// ---------------------------------------------------------------------------
// Row L2-normalize, emit f16 features
// ---------------------------------------------------------------------------
__global__ __launch_bounds__(256)
void rownorm_to_f16(const float* __restrict__ o, _Float16* __restrict__ f, int n) {
  __shared__ float red[256];
  int row = blockIdx.x, tid = threadIdx.x;
  const float* pr = o + (size_t)row * n;
  float ss = 0.f;
  for (int c = tid; c < n; c += 256) { float v = pr[c]; ss += v * v; }
  red[tid] = ss; __syncthreads();
  for (int s = 128; s > 0; s >>= 1) { if (tid < s) red[tid] += red[tid + s]; __syncthreads(); }
  float inv = 1.f / fmaxf(sqrtf(red[0]), 1e-12f);
  _Float16* pf = f + (size_t)row * n;
  for (int c = tid; c < n; c += 256) pf[c] = (_Float16)(pr[c] * inv);
}

// ---------------------------------------------------------------------------
// kNN(k=1): y[row] = argmax_j (S[row,j] - 2*[j==row]), first-max tie-break
// ---------------------------------------------------------------------------
__global__ __launch_bounds__(256)
void knn_argmax(const float* __restrict__ S, int* __restrict__ y, int n) {
  __shared__ float sv[256];
  __shared__ int   si[256];
  int row = blockIdx.x, tid = threadIdx.x;
  float best = -3.402823e38f; int bidx = n;
  for (int j = tid; j < n; j += 256) {          // in-thread j increasing -> first max kept
    float v = S[(size_t)row * n + j];
    if (j == row) v -= 2.f;
    if (v > best) { best = v; bidx = j; }
  }
  sv[tid] = best; si[tid] = bidx; __syncthreads();
  for (int s = 128; s > 0; s >>= 1) {
    if (tid < s) {
      if (sv[tid + s] > sv[tid] || (sv[tid + s] == sv[tid] && si[tid + s] < si[tid])) {
        sv[tid] = sv[tid + s]; si[tid] = si[tid + s];
      }
    }
    __syncthreads();
  }
  if (tid == 0) y[row] = si[0];
}

// ---------------------------------------------------------------------------
// Connected components via min-label propagation + pointer jumping.
// Single workgroup (1024 threads, n=2048), labels entirely in LDS.
// ---------------------------------------------------------------------------
__global__ __launch_bounds__(1024)
void cc_pointer_jump(const int* __restrict__ y, int* __restrict__ labels, int n) {
  __shared__ int sl[2048], sln[2048], sy[2048];
  int tid = threadIdx.x;
  for (int i = tid; i < n; i += 1024) { sl[i] = i; sy[i] = y[i]; }
  __syncthreads();
  for (int it = 0; it < 32; ++it) {
    for (int i = tid; i < n; i += 1024) sln[i] = min(sl[i], sl[sy[i]]);   // forward edge
    __syncthreads();
    for (int i = tid; i < n; i += 1024) atomicMin(&sln[sy[i]], sl[i]);    // reverse edge
    __syncthreads();
    int t0 = min(sln[tid],        sln[sln[tid]]);                          // pointer jumping
    int t1 = min(sln[tid + 1024], sln[sln[tid + 1024]]);
    __syncthreads();
    sl[tid] = t0; sl[tid + 1024] = t1;
    __syncthreads();
  }
  for (int i = tid; i < n; i += 1024) labels[i] = sl[i];
}

// ---------------------------------------------------------------------------
// Supervised contrastive per-row term -> rowloss[row] (deterministic).
// ---------------------------------------------------------------------------
__global__ __launch_bounds__(256)
void sup_contra_rows(const float* __restrict__ S, const int* __restrict__ labels,
                     float* __restrict__ rowloss, int n, float invT) {
  __shared__ float r0[256], r1[256], r2[256];
  int row = blockIdx.x, tid = threadIdx.x;
  int li = labels[row];
  const float* pr = S + (size_t)row * n;
  float se = 0.f, sm = 0.f, sml = 0.f;
  for (int j = tid; j < n; j += 256) {
    if (j == row) continue;
    float lg = pr[j] * invT;
    se += expf(lg);
    if (labels[j] == li) { sm += 1.f; sml += lg; }
  }
  r0[tid] = se; r1[tid] = sm; r2[tid] = sml; __syncthreads();
  for (int s = 128; s > 0; s >>= 1) {
    if (tid < s) { r0[tid] += r0[tid + s]; r1[tid] += r1[tid + s]; r2[tid] += r2[tid + s]; }
    __syncthreads();
  }
  if (tid == 0) rowloss[row] = -(r2[0] / r1[0] - logf(r0[0]));
}

__global__ __launch_bounds__(1024)
void reduce_loss(const float* __restrict__ rowloss, float* __restrict__ out,
                 int n, float scale) {
  __shared__ float red[1024];
  int tid = threadIdx.x;
  float s = 0.f;
  for (int i = tid; i < n; i += 1024) s += rowloss[i];
  red[tid] = s; __syncthreads();
  for (int st = 512; st > 0; st >>= 1) { if (tid < st) red[tid] += red[tid + st]; __syncthreads(); }
  if (tid == 0) out[0] = red[0] * scale;
}

// ---------------------------------------------------------------------------
// Orchestration
// ---------------------------------------------------------------------------
extern "C" void kernel_launch(void* const* d_in, const int* in_sizes, int n_in,
                              void* d_out, int out_size, void* d_ws, size_t ws_size,
                              hipStream_t stream) {
  (void)in_sizes; (void)n_in; (void)out_size; (void)ws_size;
  const int kB = 2048, kD = 2048, kH = 2048;

  const float* xs[2]  = {(const float*)d_in[0], (const float*)d_in[1]};
  const float* W1     = (const float*)d_in[2];
  const float* b1     = (const float*)d_in[3];
  const float* gamma  = (const float*)d_in[4];
  const float* beta   = (const float*)d_in[5];
  const float* W2     = (const float*)d_in[6];
  const float* b2     = (const float*)d_in[7];
  float* out = (float*)d_out;

  char* w = (char*)d_ws;
  auto carve = [&](size_t bytes) -> char* {
    char* p = w; w += (bytes + 255) & ~(size_t)255; return p;
  };
  _Float16* W1h  = (_Float16*)carve((size_t)kH * kD * 2);
  _Float16* W2h  = (_Float16*)carve((size_t)kD * kH * 2);
  _Float16* xh   = (_Float16*)carve((size_t)kB * kD * 2);
  float*    hf   = (float*)   carve((size_t)kB * kH * 4);   // reused: h, then o
  _Float16* hbh  = (_Float16*)carve((size_t)kB * kH * 2);
  _Float16* fth  = (_Float16*)carve((size_t)kB * kD * 2);
  float*    mu   = (float*)   carve((size_t)kH * 4);
  float*    rstd = (float*)   carve((size_t)kH * 4);
  float*    ps   = (float*)   carve((size_t)8 * kH * 4);
  float*    pss  = (float*)   carve((size_t)8 * kH * 4);
  float*    S0   = (float*)   carve((size_t)kB * kB * 4);
  float*    S1   = (float*)   carve((size_t)kB * kB * 4);
  int*      yb   = (int*)     carve((size_t)kB * 4);
  int*      l0   = (int*)     carve((size_t)kB * 4);
  int*      l1   = (int*)     carve((size_t)kB * 4);
  float*    rls  = (float*)   carve((size_t)2 * kB * 4);
  float* Sb[2]  = {S0, S1};
  int*   lab[2] = {l0, l1};

  const size_t q4 = (size_t)kB * kD / 4;            // all big tensors are 2048x2048
  const int cvtBlocks = (int)((q4 + 255) / 256);
  const dim3 ggemm(16, 8);                          // block tile 256x128 over 2048x2048
  const dim3 gbn(kH / 256, 8);                      // columns x row-chunks

  cvt_f32_to_f16<<<cvtBlocks, 256, 0, stream>>>(W1, W1h, q4);
  cvt_f32_to_f16<<<cvtBlocks, 256, 0, stream>>>(W2, W2h, q4);

  for (int p = 0; p < 2; ++p) {
    cvt_f32_to_f16<<<cvtBlocks, 256, 0, stream>>>(xs[p], xh, q4);
    gemm_nt_wmma<<<ggemm, 256, 0, stream>>>(xh, W1h, hf, b1, kB, kH, kD);   // h = x@W1^T + b1
    bn_partial<<<gbn, 256, 0, stream>>>(hf, ps, pss, kH);
    bn_finalize<<<kH / 256, 256, 0, stream>>>(ps, pss, mu, rstd, 8, kB, kH);
    bn_relu_f16<<<cvtBlocks, 256, 0, stream>>>(hf, mu, rstd, gamma, beta, hbh, kH, q4);
    gemm_nt_wmma<<<ggemm, 256, 0, stream>>>(hbh, W2h, hf, b2, kB, kD, kH);  // o = relu(bn)@W2^T + b2
    rownorm_to_f16<<<kB, 256, 0, stream>>>(hf, fth, kD);                    // feat (f16)
    gemm_nt_wmma<<<ggemm, 256, 0, stream>>>(fth, fth, Sb[p], nullptr, kB, kB, kD); // S = feat@feat^T
    knn_argmax<<<kB, 256, 0, stream>>>(Sb[p], yb, kB);
    cc_pointer_jump<<<1, 1024, 0, stream>>>(yb, lab[p], kB);
  }

  const float invT = 10.f;                          // 1/T
  sup_contra_rows<<<kB, 256, 0, stream>>>(Sb[0], lab[1], rls,      kB, invT);
  sup_contra_rows<<<kB, 256, 0, stream>>>(Sb[1], lab[0], rls + kB, kB, invT);
  reduce_loss<<<1, 1024, 0, stream>>>(rls, out, 2 * kB, 1.f / (2.f * kB));
}
